// MultiHeadAttention_20787641712887
// MI455X (gfx1250) — matrix operand order repaired and checked by reading, not compiled, not run on hardware
//
#include <hip/hip_runtime.h>
#include <hip/hip_bf16.h>
#include <type_traits>

typedef __bf16 bf16_t;
typedef __attribute__((ext_vector_type(16))) __bf16 v16bf;
typedef __attribute__((ext_vector_type(8)))  __bf16 v8bf;
typedef __attribute__((ext_vector_type(8)))  float  v8f;
typedef __attribute__((ext_vector_type(4)))  unsigned int v4u;
typedef __attribute__((ext_vector_type(8)))  int v8i;
typedef __attribute__((ext_vector_type(4)))  int v4i;

#define N_HEADC 8
#define D_HEADC 64
#define D_MODELC 512
#define SEQ_L 1024
#define BATCHC 8

#if defined(__has_builtin)
#if __has_builtin(__builtin_amdgcn_tensor_load_to_lds)
#define HAVE_TDM 1
#endif
#endif
#ifndef HAVE_TDM
#define HAVE_TDM 0
#endif

__device__ __forceinline__ bf16_t f2bf(float f) { return (bf16_t)f; }

// A-matrix 16x32 bf16 fragment K index (ISA 7.12.2): lane-halves interleave K
// in 8-wide chunks: lanes 0-15 hold K {0..7, 16..23}, lanes 16-31 {8..15, 24..31}.
__device__ __forceinline__ int frag_ka(int p, int half, int j) {
    return 16 * (p >> 2) + 8 * half + 2 * (p & 3) + j;
}
// B-matrix 32x16 bf16 fragment K index: same order as A: lanes 0-15 hold
// K {0..7, 16..23}, lanes 16-31 hold K {8..15, 24..31}.
__device__ __forceinline__ int frag_kb(int p, int half, int j) {
    return 16 * (p >> 2) + 8 * half + 2 * (p & 3) + j;
}

template <typename F>
__device__ __forceinline__ v16bf load_a_frag(F elem) {
    const int lane = threadIdx.x & 31;
    const int m = lane & 15, half = lane >> 4;
    v16bf a;
#pragma unroll
    for (int p = 0; p < 8; ++p) {
        a[2 * p]     = elem(m, frag_ka(p, half, 0));
        a[2 * p + 1] = elem(m, frag_ka(p, half, 1));
    }
    return a;
}

template <typename F>
__device__ __forceinline__ v16bf load_b_frag(F elem) {
    const int lane = threadIdx.x & 31;
    const int n = lane & 15, half = lane >> 4;
    v16bf b;
#pragma unroll
    for (int p = 0; p < 8; ++p) {
        b[2 * p]     = elem(frag_kb(p, half, 0), n);
        b[2 * p + 1] = elem(frag_kb(p, half, 1), n);
    }
    return b;
}

__device__ __forceinline__ v8f wmma_bf16(v16bf a, v16bf b, v8f c) {
    return __builtin_amdgcn_wmma_f32_16x16x32_bf16(false, a, false, b, (short)0, c,
                                                   false, false);
}

// CDNA5 async DMA: global -> LDS, 16B per lane, tracked by ASYNCcnt.
__device__ __forceinline__ void async_copy_b128(unsigned int lds_addr,
                                                const void* gptr) {
    asm volatile("global_load_async_to_lds_b128 %0, %1, off"
                 :: "v"(lds_addr),
                    "v"((unsigned long long)(uintptr_t)gptr)
                 : "memory");
}
__device__ __forceinline__ void wait_async0() {
    asm volatile("s_wait_asynccnt 0x0" ::: "memory");
}

#if HAVE_TDM
// Tensor Data Mover: 1-D copy of nelem 2-byte elements, global -> LDS.
// D# group 0 (ISA 8.3): count=1 | lds_addr[63:32] | global_addr[120:64] | type=2.
// D# group 1 (ISA 8.4): data_size=1 (2B); tensor_dim0 = tile_dim0 = nelem;
// tensor_dim1 = tile_dim1 = 1; strides = nelem.
__device__ __forceinline__ void tdm_load_1d(unsigned int lds_addr,
                                            const void* gptr,
                                            unsigned int nelem) {
    unsigned long long ga = (unsigned long long)(uintptr_t)gptr;
    v4u g0;
    g0.x = 1u;                                            // count=1
    g0.y = lds_addr;                                      // LDS byte address
    g0.z = (unsigned int)(ga & 0xFFFFFFFFu);              // global_addr[31:0]
    g0.w = (unsigned int)((ga >> 32) & 0x01FFFFFFu) | (2u << 30);  // type=2
    v8i g1;
    g1[0] = (int)(1u << 16);                              // data_size = 2 bytes
    g1[1] = (int)((nelem & 0xFFFFu) << 16);               // tensor_dim0[15:0]
    g1[2] = (int)((nelem >> 16) | (1u << 16));            // dim0 hi | tensor_dim1=1
    g1[3] = (int)((nelem & 0xFFFFu) << 16);               // tile_dim0
    g1[4] = 1;                                            // tile_dim1=1, tile_dim2=0
    g1[5] = (int)nelem;                                   // tensor_dim0_stride lo
    g1[6] = (int)((nelem & 0xFFFFu) << 16);               // s0 hi=0 | s1[15:0]
    g1[7] = (int)(nelem >> 16);                           // tensor_dim1_stride hi
    v4i z = {0, 0, 0, 0};
#if __clang_major__ >= 23
    v8i z8 = {0, 0, 0, 0, 0, 0, 0, 0};
    __builtin_amdgcn_tensor_load_to_lds(g0, g1, z, z, z8, 0);
#else
    __builtin_amdgcn_tensor_load_to_lds(g0, g1, z, z, 0);
#endif
}
#endif

// C = X[M,K] @ W[K,N] + bias.  STORE_BHLD=1: scatter bf16 to [B,H,L,D] layout
// (rows are (l,b) pairs of the [L,B,512] activations). STORE_BHLD=0: dense f32.
template <typename TIN, int STORE_BHLD>
__global__ __launch_bounds__(128)
void gemm_bias_kernel(const TIN* __restrict__ X, const float* __restrict__ W,
                      const float* __restrict__ bias, float* __restrict__ outF,
                      bf16_t* __restrict__ outB, int M, int N, int K) {
    __shared__ __align__(16) bf16_t Xs[64][40];  // A tile [m][k]
    __shared__ __align__(16) bf16_t Wt[64][40];  // B tile transposed: [n][k]
    const int tid  = threadIdx.x;
    const int wave = tid >> 5;
    const int lane = tid & 31;
    const int half = lane >> 4;
    const int m0 = blockIdx.y * 64;
    const int n0 = blockIdx.x * 64;
    const unsigned int xsBase = (unsigned int)(uintptr_t)&Xs[0][0];

    v8f acc[4] = {};
    for (int k0 = 0; k0 < K; k0 += 32) {
        __syncthreads();
        // Stage X tile (64x32) as bf16.
        if constexpr (std::is_same<TIN, float>::value) {
#pragma unroll
            for (int j = 0; j < 4; ++j) {
                int chunk = tid + 128 * j;               // 512 chunks of 4 f32
                int r = chunk >> 3, c = (chunk & 7) * 4;
                const float4 x =
                    *(const float4*)&X[(size_t)(m0 + r) * K + k0 + c];
                Xs[r][c + 0] = f2bf(x.x);
                Xs[r][c + 1] = f2bf(x.y);
                Xs[r][c + 2] = f2bf(x.z);
                Xs[r][c + 3] = f2bf(x.w);
            }
        } else {
            // bf16 input needs no conversion: async DMA straight into LDS.
#pragma unroll
            for (int j = 0; j < 2; ++j) {
                int chunk = tid + 128 * j;               // 256 chunks of 8 bf16
                int r = chunk >> 2, c = (chunk & 3) * 8;
                async_copy_b128(xsBase + (unsigned int)(r * 40 + c) * 2,
                                &X[(size_t)(m0 + r) * K + k0 + c]);
            }
        }
        // Stage W tile (32x64) transposed into [n][k].
#pragma unroll
        for (int j = 0; j < 4; ++j) {
            int chunk = tid + 128 * j;                   // 512 chunks of 4 f32
            int r = chunk >> 4, c = (chunk & 15) * 4;
            const float4 w = *(const float4*)&W[(size_t)(k0 + r) * N + n0 + c];
            Wt[c + 0][r] = f2bf(w.x);
            Wt[c + 1][r] = f2bf(w.y);
            Wt[c + 2][r] = f2bf(w.z);
            Wt[c + 3][r] = f2bf(w.w);
        }
        if constexpr (!std::is_same<TIN, float>::value) wait_async0();
        __syncthreads();
        v16bf a = load_a_frag([&](int m, int k) { return Xs[wave * 16 + m][k]; });
#pragma unroll
        for (int nt = 0; nt < 4; ++nt) {
            v16bf b = load_b_frag([&](int k, int n) { return Wt[nt * 16 + n][k]; });
            acc[nt] = wmma_bf16(a, b, acc[nt]);
        }
    }

#pragma unroll
    for (int nt = 0; nt < 4; ++nt) {
#pragma unroll
        for (int r = 0; r < 8; ++r) {
            int m = m0 + wave * 16 + r + 8 * half;
            int n = n0 + nt * 16 + (lane & 15);
            float v = acc[nt][r] + bias[n];
            if (STORE_BHLD) {
                int l = m / BATCHC, b = m % BATCHC;
                int h = n / D_HEADC, d = n % D_HEADC;
                outB[(((size_t)b * N_HEADC + h) * SEQ_L + l) * D_HEADC + d] =
                    f2bf(v);
            } else {
                outF[(size_t)m * N + n] = v;
            }
        }
    }
}

// Flash attention: block = 64 queries of one (b,h); 4 waves x 16 queries.
__global__ __launch_bounds__(128)
void attn_kernel(const bf16_t* __restrict__ qg, const bf16_t* __restrict__ kg,
                 const bf16_t* __restrict__ vg, const float* __restrict__ mask,
                 bf16_t* __restrict__ ao) {
    __shared__ __align__(16) bf16_t Ks[64][64];   // [key][d], TDM/DMA target
    __shared__ __align__(16) bf16_t Vt[64][72];   // [d][key] transposed
    __shared__ __align__(16) bf16_t Ps[4][16][72];

    const int tid  = threadIdx.x;
    const int wave = tid >> 5;
    const int lane = tid & 31;
    const int half = lane >> 4;
    const int bh = blockIdx.y;
    const int b = bh / N_HEADC, h = bh % N_HEADC;
    const int q0 = blockIdx.x * 64;
    const size_t base = ((size_t)b * N_HEADC + h) * SEQ_L;
    const int qRow0 = q0 + wave * 16;
    const unsigned int ksBase = (unsigned int)(uintptr_t)&Ks[0][0];

    v16bf aq0 = load_a_frag(
        [&](int m, int k) { return qg[(base + qRow0 + m) * D_HEADC + k]; });
    v16bf aq1 = load_a_frag(
        [&](int m, int k) { return qg[(base + qRow0 + m) * D_HEADC + 32 + k]; });

    float rowMax[8], rowSum[8];
    v8f o[4] = {};
#pragma unroll
    for (int r = 0; r < 8; ++r) { rowMax[r] = -3.0e38f; rowSum[r] = 0.0f; }

    for (int kb = 0; kb < SEQ_L; kb += 64) {
        __syncthreads();
        // K tile: contiguous 8 KB -> LDS.  TDM if available, else async DMA.
        const bf16_t* ksrc = kg + (base + kb) * D_HEADC;
#if HAVE_TDM
        if (tid < 32) tdm_load_1d(ksBase, ksrc, 64 * 64);
#else
#pragma unroll
        for (int j = 0; j < 4; ++j) {
            int chunk = tid + 128 * j;                    // 512 x 16B
            async_copy_b128(ksBase + chunk * 16, ksrc + chunk * 8);
        }
#endif
        // V tile: b128 loads, transposed stores into [d][key].
#pragma unroll
        for (int j = 0; j < 4; ++j) {
            int chunk = tid + 128 * j;                    // 512 chunks of 8 bf16
            int r = chunk >> 3, c = (chunk & 7) * 8;
            v8bf v = *(const v8bf*)&vg[(base + kb + r) * D_HEADC + c];
#pragma unroll
            for (int e = 0; e < 8; ++e) Vt[c + e][r] = v[e];
        }
        if (kb + 64 < SEQ_L) {  // prefetch next tiles toward L2
            __builtin_prefetch(kg + (base + kb + 64) * D_HEADC, 0, 1);
            __builtin_prefetch(vg + (base + kb + 64) * D_HEADC, 0, 1);
        }
#if HAVE_TDM
        if (tid < 32) __builtin_amdgcn_s_wait_tensorcnt(0);
#else
        wait_async0();
#endif
        __syncthreads();

        // S = Q K^T  (B-fragment reads K^T: per-lane contiguous d run)
        v8f s[4] = {};
#pragma unroll
        for (int nt = 0; nt < 4; ++nt) {
            v16bf b0 = load_b_frag(
                [&](int k, int n) { return Ks[nt * 16 + n][k]; });
            s[nt] = wmma_bf16(aq0, b0, s[nt]);
            v16bf b1 = load_b_frag(
                [&](int k, int n) { return Ks[nt * 16 + n][32 + k]; });
            s[nt] = wmma_bf16(aq1, b1, s[nt]);
        }

        // Online softmax.  Accumulator row m = r + 8*half, col = nt*16 + lane&15.
        float sv[4][8], bmax[8], bsum[8];
#pragma unroll
        for (int r = 0; r < 8; ++r) bmax[r] = -3.0e38f;
#pragma unroll
        for (int nt = 0; nt < 4; ++nt) {
            int key = kb + nt * 16 + (lane & 15);
#pragma unroll
            for (int r = 0; r < 8; ++r) {
                int q = q0 + wave * 16 + r + 8 * half;
                float v = s[nt][r] * 0.125f + mask[(size_t)q * SEQ_L + key];
                sv[nt][r] = v;
                bmax[r] = fmaxf(bmax[r], v);
            }
        }
#pragma unroll
        for (int r = 0; r < 8; ++r)
#pragma unroll
            for (int off = 8; off > 0; off >>= 1)
                bmax[r] = fmaxf(bmax[r], __shfl_xor(bmax[r], off, 16));

        float corr[8], nmax[8];
#pragma unroll
        for (int r = 0; r < 8; ++r) {
            nmax[r] = fmaxf(rowMax[r], bmax[r]);
            corr[r] = __expf(rowMax[r] - nmax[r]);
            rowMax[r] = nmax[r];
            bsum[r] = 0.0f;
        }
#pragma unroll
        for (int nt = 0; nt < 4; ++nt) {
#pragma unroll
            for (int r = 0; r < 8; ++r) {
                float p = __expf(sv[nt][r] - nmax[r]);
                bsum[r] += p;
                Ps[wave][r + 8 * half][nt * 16 + (lane & 15)] = f2bf(p);
            }
        }
#pragma unroll
        for (int r = 0; r < 8; ++r) {
#pragma unroll
            for (int off = 8; off > 0; off >>= 1)
                bsum[r] += __shfl_xor(bsum[r], off, 16);
            rowSum[r] = rowSum[r] * corr[r] + bsum[r];
#pragma unroll
            for (int nt = 0; nt < 4; ++nt) o[nt][r] *= corr[r];
        }
        __syncthreads();  // Ps visible for A-fragment reads

        // O += P V   (B-fragment reads V^T: per-lane contiguous key run)
        v16bf ap0 = load_a_frag([&](int m, int k) { return Ps[wave][m][k]; });
        v16bf ap1 = load_a_frag([&](int m, int k) { return Ps[wave][m][32 + k]; });
#pragma unroll
        for (int nt = 0; nt < 4; ++nt) {
            v16bf bv0 = load_b_frag(
                [&](int k, int n) { return Vt[nt * 16 + n][k]; });
            o[nt] = wmma_bf16(ap0, bv0, o[nt]);
            v16bf bv1 = load_b_frag(
                [&](int k, int n) { return Vt[nt * 16 + n][32 + k]; });
            o[nt] = wmma_bf16(ap1, bv1, o[nt]);
        }
    }

    // Store O / rowSum as bf16 rows of the [L*B, 512] matrix for the out-proj.
#pragma unroll
    for (int nt = 0; nt < 4; ++nt) {
#pragma unroll
        for (int r = 0; r < 8; ++r) {
            int l = q0 + wave * 16 + r + 8 * half;
            int d = nt * 16 + (lane & 15);
            ao[((size_t)l * BATCHC + b) * D_MODELC + h * D_HEADC + d] =
                f2bf(o[nt][r] / rowSum[r]);
        }
    }
}

extern "C" void kernel_launch(void* const* d_in, const int* in_sizes, int n_in,
                              void* d_out, int out_size, void* d_ws, size_t ws_size,
                              hipStream_t stream) {
    (void)in_sizes; (void)n_in; (void)out_size; (void)ws_size;
    const float* Q    = (const float*)d_in[0];
    const float* K    = (const float*)d_in[1];
    const float* V    = (const float*)d_in[2];
    const float* mask = (const float*)d_in[3];
    const float* Wq = (const float*)d_in[4];
    const float* bq = (const float*)d_in[5];
    const float* Wk = (const float*)d_in[6];
    const float* bk = (const float*)d_in[7];
    const float* Wv = (const float*)d_in[8];
    const float* bv = (const float*)d_in[9];
    const float* Wo = (const float*)d_in[10];
    const float* bo = (const float*)d_in[11];
    float* out = (float*)d_out;

    const int M = SEQ_L * BATCHC;  // 8192 rows
    const size_t qkvElems = (size_t)BATCHC * N_HEADC * SEQ_L * D_HEADC;  // 4 Mi
    bf16_t* qg = (bf16_t*)d_ws;
    bf16_t* kg = qg + qkvElems;
    bf16_t* vg = kg + qkvElems;
    bf16_t* ao = vg + qkvElems;  // 8192*512 bf16; total ws = 32 MB

    dim3 blk(128);
    dim3 gGrid(D_MODELC / 64, M / 64);
    gemm_bias_kernel<float, 1><<<gGrid, blk, 0, stream>>>(Q, Wq, bq, nullptr, qg,
                                                          M, D_MODELC, D_MODELC);
    gemm_bias_kernel<float, 1><<<gGrid, blk, 0, stream>>>(K, Wk, bk, nullptr, kg,
                                                          M, D_MODELC, D_MODELC);
    gemm_bias_kernel<float, 1><<<gGrid, blk, 0, stream>>>(V, Wv, bv, nullptr, vg,
                                                          M, D_MODELC, D_MODELC);

    dim3 aGrid(SEQ_L / 64, BATCHC * N_HEADC);
    attn_kernel<<<aGrid, blk, 0, stream>>>(qg, kg, vg, mask, ao);

    gemm_bias_kernel<bf16_t, 0><<<gGrid, blk, 0, stream>>>(ao, Wo, bo, out, nullptr,
                                                           M, D_MODELC, D_MODELC);
}
